// ICLAttentionAlt_52493090292405
// MI455X (gfx1250) — matrix operand order repaired and checked by reading, not compiled, not run on hardware
//
#include <hip/hip_runtime.h>

#define HEADS 8
#define DE    64
#define BATCH 2
#define SEQ   2048
#define BH    (BATCH * HEADS)         // 16
#define ZDIM  (HEADS * DE)            // 512
#define ROWS  (BATCH * SEQ)           // 4096
#define NEG_BIG (-3.0e38f)

typedef __attribute__((ext_vector_type(16))) __bf16 v16bf;
typedef __attribute__((ext_vector_type(8)))  float  v8f;

union BF8 { uint4 u4; __bf16 b[8]; };

__device__ __forceinline__ int lane_id() { return (int)(threadIdx.x & 31u); }

__device__ __forceinline__ v8f v8f_zero() {
  v8f z = {0.f, 0.f, 0.f, 0.f, 0.f, 0.f, 0.f, 0.f};
  return z;
}

__device__ __forceinline__ v8f wmma_bf16(v16bf a, v16bf b, v8f c) {
  // v_wmma_f32_16x16x32_bf16: D = A(16x32) * B(32x16) + C(16x16 f32)
  return __builtin_amdgcn_wmma_f32_16x16x32_bf16(false, a, false, b, (short)0, c,
                                                 false, false);
}

// ---- A fragment (16x32, 16-bit): lane = {row = l&15, half = l>>4}.
// Per ISA: VGPR0..3 hold K = half*8 + 0..7, VGPR4..7 hold K = 16 + half*8 + 0..7
// -> two contiguous 8-element runs per lane.
__device__ __forceinline__ v16bf load_a_bf16(const __bf16* base, int ld, int kbase) {
  const int l = lane_id();
  const __bf16* p = base + (size_t)(l & 15) * ld + kbase + (l >> 4) * 8;
  BF8 lo, hi;
  lo.u4 = *(const uint4*)(p);
  hi.u4 = *(const uint4*)(p + 16);
  v16bf a;
#pragma unroll
  for (int i = 0; i < 8; ++i) { a[i] = lo.b[i]; a[i + 8] = hi.b[i]; }
  return a;
}

__device__ __forceinline__ v16bf load_a_f32cvt(const float* base, int ld, int kbase) {
  const int l = lane_id();
  const float* p = base + (size_t)(l & 15) * ld + kbase + (l >> 4) * 8;
  v16bf a;
#pragma unroll
  for (int i = 0; i < 8; ++i) { a[i] = (__bf16)p[i]; a[i + 8] = (__bf16)p[16 + i]; }
  return a;
}

// ---- B fragment (32x16, 16-bit): lane = {col N = l&15, half = l>>4},
// elements 0..7 = K half*8 + 0..7, elements 8..15 = K 16 + half*8 + 0..7
// (same K order as A; strided by ld in a row-major [K][N] source).
__device__ __forceinline__ v16bf load_b_bf16(const __bf16* base, int ld) {
  const __bf16* p = base + (size_t)((lane_id() >> 4) * 8) * ld + (lane_id() & 15);
  v16bf b;
#pragma unroll
  for (int i = 0; i < 8; ++i) { b[i] = p[(size_t)i * ld]; b[i + 8] = p[(size_t)(16 + i) * ld]; }
  return b;
}

__device__ __forceinline__ v16bf load_b_f32cvt(const float* base, int ld) {
  const float* p = base + (size_t)((lane_id() >> 4) * 8) * ld + (lane_id() & 15);
  v16bf b;
#pragma unroll
  for (int i = 0; i < 8; ++i) { b[i] = (__bf16)p[(size_t)i * ld]; b[i + 8] = (__bf16)p[(size_t)(16 + i) * ld]; }
  return b;
}

// =====================================================================
// Kernel 1: qh/kh/vh head projections (bf16 WMMA) + RoPE on q,k.
// One wave per (matrix, head, 16-row tile). Stores:
//   qh  [bh][s][d]  bf16 (row major)   -> A frags for score GEMMs
//   khT [bh][d][s]  bf16 (d major)     -> B frags for score GEMMs
//   vh  [bh][s][d]  bf16 (row major)   -> B frags for P^T * V
// =====================================================================
__global__ __launch_bounds__(128) void k_proj_rope(
    const float* __restrict__ q, const float* __restrict__ k,
    const float* __restrict__ v, const float* __restrict__ Wq,
    const float* __restrict__ Wk, const float* __restrict__ Wv,
    __bf16* __restrict__ qh, __bf16* __restrict__ khT, __bf16* __restrict__ vh) {
  const int wave = blockIdx.x * (blockDim.x >> 5) + (threadIdx.x >> 5);
  const int tilesPerHead = ROWS / 16;                 // 256
  const int mat  = wave / (HEADS * tilesPerHead);     // 0=q 1=k 2=v
  const int rem  = wave % (HEADS * tilesPerHead);
  const int h    = rem / tilesPerHead;
  const int tile = rem % tilesPerHead;
  const int row0 = tile * 16;                         // row in [0, B*S)

  const float* X = (mat == 0) ? q : (mat == 1) ? k : v;
  const float* W = ((mat == 0) ? Wq : (mat == 1) ? Wk : Wv) + (size_t)h * DE * DE;

  v16bf a0 = load_a_f32cvt(X + (size_t)row0 * DE, DE, 0);
  v16bf a1 = load_a_f32cvt(X + (size_t)row0 * DE, DE, 32);

  v8f acc[4];
#pragma unroll
  for (int n = 0; n < 4; ++n) acc[n] = v8f_zero();
#pragma unroll
  for (int c = 0; c < 2; ++c) {
    v16bf a = c ? a1 : a0;
#pragma unroll
    for (int n = 0; n < 4; ++n) {
      v16bf b = load_b_f32cvt(W + (size_t)(c * 32) * DE + n * 16, DE);
      acc[n] = wmma_bf16(a, b, acc[n]);
    }
  }

  const int l = lane_id(), nlo = l & 15, hs = l >> 4;

  if (mat < 2) {  // RoPE: pairs are adjacent columns -> adjacent lanes in C tile
#pragma unroll
    for (int n = 0; n < 4; ++n) {
      const int N = n * 16 + nlo;
      const float inv_freq = __powf(10000.0f, -(float)(2 * (N >> 1)) / 64.0f);
#pragma unroll
      for (int r = 0; r < 8; ++r) {
        const int spos = (row0 + r + 8 * hs) & (SEQ - 1);
        float sn, cs;
        __sincosf((float)spos * inv_freq, &sn, &cs);
        float mine  = acc[n][r];
        float other = __shfl_xor(mine, 1, 32);
        // even col: x1*cos - x2*sin ; odd col: x1*sin + x2*cos
        acc[n][r] = ((N & 1) == 0) ? (mine * cs - other * sn)
                                   : (other * sn + mine * cs);
      }
    }
  }

  const int b  = row0 / SEQ;
  const int sr = row0 % SEQ;
  const int bh = b * HEADS + h;

  if (mat == 1) {  // k -> transposed layout [bh][d][s], packed 8-wide stores
#pragma unroll
    for (int n = 0; n < 4; ++n) {
      const int d = n * 16 + nlo;
      __bf16* dst = khT + ((size_t)bh * DE + d) * SEQ + sr + 8 * hs;
      BF8 pk;
#pragma unroll
      for (int r = 0; r < 8; ++r) pk.b[r] = (__bf16)acc[n][r];
      *(uint4*)dst = pk.u4;
    }
  } else {
    __bf16* dst = ((mat == 0) ? qh : vh) + ((size_t)bh * SEQ + sr) * DE;
#pragma unroll
    for (int n = 0; n < 4; ++n)
#pragma unroll
      for (int r = 0; r < 8; ++r)
        dst[(size_t)(r + 8 * hs) * DE + n * 16 + nlo] = (__bf16)acc[n][r];
  }
}

// =====================================================================
// Kernel 2: per-row softmax statistics m[q], l[q] (online max/sum over
// causal k-tiles). One wave per (bh, q-tile).
// =====================================================================
__global__ __launch_bounds__(128) void k_softmax_stats(
    const __bf16* __restrict__ qh, const __bf16* __restrict__ khT,
    float* __restrict__ m_arr, float* __restrict__ l_arr) {
  const int wave = blockIdx.x * (blockDim.x >> 5) + (threadIdx.x >> 5);
  const int bh = wave / (SEQ / 16);
  const int qt = wave % (SEQ / 16);

  const __bf16* Q  = qh + ((size_t)bh * SEQ + qt * 16) * DE;
  const __bf16* KT = khT + (size_t)bh * DE * SEQ;
  v16bf aq0 = load_a_bf16(Q, DE, 0);
  v16bf aq1 = load_a_bf16(Q, DE, 32);

  const int l = lane_id(), nlo = l & 15, hs = l >> 4;
  float mrow[8], lrow[8];
#pragma unroll
  for (int r = 0; r < 8; ++r) { mrow[r] = NEG_BIG; lrow[r] = 0.f; }

  for (int kt = 0; kt <= qt; ++kt) {
    v8f s = v8f_zero();
    s = wmma_bf16(aq0, load_b_bf16(KT + kt * 16, SEQ), s);
    s = wmma_bf16(aq1, load_b_bf16(KT + (size_t)32 * SEQ + kt * 16, SEQ), s);
#pragma unroll
    for (int r = 0; r < 8; ++r) {
      const int qg = qt * 16 + r + 8 * hs;
      const int kg = kt * 16 + nlo;
      float x = (kg <= qg) ? s[r] * (1.0f / 64.0f) : NEG_BIG;
      float gm = x;  // row max across the 16 lanes of this half
#pragma unroll
      for (int off = 1; off < 16; off <<= 1) gm = fmaxf(gm, __shfl_xor(gm, off, 32));
      const float mn = fmaxf(mrow[r], gm);
      float e = __expf(x - mn);
#pragma unroll
      for (int off = 1; off < 16; off <<= 1) e += __shfl_xor(e, off, 32);
      lrow[r] = lrow[r] * __expf(mrow[r] - mn) + e;
      mrow[r] = mn;
    }
  }
  if (nlo == 0) {
#pragma unroll
    for (int r = 0; r < 8; ++r) {
      const int idx = bh * SEQ + qt * 16 + r + 8 * hs;
      m_arr[idx] = mrow[r];
      l_arr[idx] = lrow[r];
    }
  }
}

// =====================================================================
// Kernel 3: out[bh][k][d] = sum_q P[q,k] * vh[q,d]   (transposed PV!)
// One wave owns one k-tile (16 x 64 output, register-resident), loops
// over q in steps of 32: 4 score WMMAs -> P -> LDS transpose -> 4 PtV
// WMMAs. Diagonal removal and causality: p = 0 unless k < q.
// Writes f32 to outh laid out [b*S][H*E] for the final projection.
// =====================================================================
__global__ __launch_bounds__(128) void k_attn_out(
    const __bf16* __restrict__ qh, const __bf16* __restrict__ khT,
    const __bf16* __restrict__ vh, const float* __restrict__ m_arr,
    const float* __restrict__ l_arr, float* __restrict__ outh) {
  __shared__ __attribute__((aligned(16))) __bf16 ptbuf[4][16 * 32];
  const int wave = blockIdx.x * (blockDim.x >> 5) + (threadIdx.x >> 5);
  const int bh = wave / (SEQ / 16);
  const int kt = wave % (SEQ / 16);
  __bf16* PT = &ptbuf[threadIdx.x >> 5][0];

  const __bf16* Qb = qh  + (size_t)bh * SEQ * DE;
  const __bf16* Vb = vh  + (size_t)bh * SEQ * DE;
  const __bf16* KT = khT + (size_t)bh * DE * SEQ + kt * 16;
  const float*  mb = m_arr + (size_t)bh * SEQ;
  const float*  lb = l_arr + (size_t)bh * SEQ;

  // K-tile B fragments are loop-invariant for this wave
  v16bf bk0 = load_b_bf16(KT, SEQ);
  v16bf bk1 = load_b_bf16(KT + (size_t)32 * SEQ, SEQ);

  v8f o[4];
#pragma unroll
  for (int n = 0; n < 4; ++n) o[n] = v8f_zero();

  const int l = lane_id(), nlo = l & 15, hs = l >> 4;
  const int kg = kt * 16 + nlo;  // this lane's k column in score tiles

  for (int qc = (kt >> 1) * 32; qc < SEQ; qc += 32) {
    __builtin_prefetch(Vb + (size_t)(qc + 32) * DE, 0, 0);
#pragma unroll
    for (int sub = 0; sub < 2; ++sub) {
      const int q0 = qc + sub * 16;
      v8f s = v8f_zero();
      s = wmma_bf16(load_a_bf16(Qb + (size_t)q0 * DE, DE, 0), bk0, s);
      s = wmma_bf16(load_a_bf16(Qb + (size_t)q0 * DE, DE, 32), bk1, s);
#pragma unroll
      for (int r = 0; r < 8; ++r) {
        const int qg = q0 + r + 8 * hs;
        float p = 0.f;
        if (kg < qg)  // causal AND zeroed diagonal => strictly k < q
          p = __expf(s[r] * (1.0f / 64.0f) - mb[qg]) / lb[qg];
        PT[nlo * 32 + sub * 16 + r + 8 * hs] = (__bf16)p;  // P^T[k][q]
      }
    }
    asm volatile("s_wait_dscnt 0" ::: "memory");  // LDS transpose visible
    v16bf ap = load_a_bf16(PT, 32, 0);            // P^T (16k x 32q)
#pragma unroll
    for (int n = 0; n < 4; ++n) {
      v16bf bv = load_b_bf16(Vb + (size_t)qc * DE + n * 16, DE);  // V (32q x 16d)
      o[n] = wmma_bf16(ap, bv, o[n]);
    }
  }

  const int b = bh / HEADS, h = bh % HEADS;
  float* dst = outh + ((size_t)b * SEQ + kt * 16) * ZDIM + h * DE;
#pragma unroll
  for (int n = 0; n < 4; ++n)
#pragma unroll
    for (int r = 0; r < 8; ++r)
      dst[(size_t)(r + 8 * hs) * ZDIM + n * 16 + nlo] = o[n][r];
}

// =====================================================================
// Kernel 4: final projection  (B*S x 512) @ (512 x 64) -> d_out f32
// One wave per 16-row s-tile; 16 K-chunks x 4 N-tiles = 64 WMMAs.
// =====================================================================
__global__ __launch_bounds__(128) void k_out_proj(
    const float* __restrict__ outh, const float* __restrict__ Wo,
    float* __restrict__ out) {
  const int wave = blockIdx.x * (blockDim.x >> 5) + (threadIdx.x >> 5);
  const float* A = outh + (size_t)wave * 16 * ZDIM;

  v8f acc[4];
#pragma unroll
  for (int n = 0; n < 4; ++n) acc[n] = v8f_zero();

  for (int c = 0; c < 16; ++c) {
    v16bf a = load_a_f32cvt(A, ZDIM, c * 32);
#pragma unroll
    for (int n = 0; n < 4; ++n) {
      v16bf b = load_b_f32cvt(Wo + (size_t)(c * 32) * DE + n * 16, DE);
      acc[n] = wmma_bf16(a, b, acc[n]);
    }
  }

  const int l = lane_id(), nlo = l & 15, hs = l >> 4;
  float* dst = out + (size_t)wave * 16 * DE;
#pragma unroll
  for (int n = 0; n < 4; ++n)
#pragma unroll
    for (int r = 0; r < 8; ++r)
      dst[(size_t)(r + 8 * hs) * DE + n * 16 + nlo] = acc[n][r];
}

extern "C" void kernel_launch(void* const* d_in, const int* in_sizes, int n_in,
                              void* d_out, int out_size, void* d_ws, size_t ws_size,
                              hipStream_t stream) {
  (void)in_sizes; (void)n_in; (void)out_size; (void)ws_size;
  const float* q  = (const float*)d_in[0];
  const float* k  = (const float*)d_in[1];
  const float* v  = (const float*)d_in[2];
  const float* Wq = (const float*)d_in[3];
  const float* Wk = (const float*)d_in[4];
  const float* Wv = (const float*)d_in[5];
  const float* Wo = (const float*)d_in[6];
  float* out = (float*)d_out;

  char* ws = (char*)d_ws;
  size_t off = 0;
  __bf16* qh  = (__bf16*)(ws + off); off += (size_t)BH * SEQ * DE * 2;   // 4 MB
  __bf16* khT = (__bf16*)(ws + off); off += (size_t)BH * SEQ * DE * 2;   // 4 MB
  __bf16* vh  = (__bf16*)(ws + off); off += (size_t)BH * SEQ * DE * 2;   // 4 MB
  float* m_arr = (float*)(ws + off); off += (size_t)BH * SEQ * 4;        // 128 KB
  float* l_arr = (float*)(ws + off); off += (size_t)BH * SEQ * 4;        // 128 KB
  float* outh  = (float*)(ws + off); off += (size_t)ROWS * ZDIM * 4;     // 8 MB

  // waves: k1 = 3*8*256 = 6144, k2 = k3 = 16*128 = 2048, k4 = 256 (4 waves/block)
  k_proj_rope   <<<1536, 128, 0, stream>>>(q, k, v, Wq, Wk, Wv, qh, khT, vh);
  k_softmax_stats<<<512, 128, 0, stream>>>(qh, khT, m_arr, l_arr);
  k_attn_out    <<<512, 128, 0, stream>>>(qh, khT, vh, m_arr, l_arr, outh);
  k_out_proj    <<< 64, 128, 0, stream>>>(outh, Wo, out);
}
